// GQAAttention_63797444215604
// MI455X (gfx1250) — matrix operand order repaired and checked by reading, not compiled, not run on hardware
//
#include <hip/hip_runtime.h>
#include <math.h>

typedef unsigned short u16;
typedef __bf16 bf16_t;
typedef bf16_t v16bf __attribute__((ext_vector_type(16)));
typedef float v8f __attribute__((ext_vector_type(8)));

union FragU { v16bf v; uint4 q[2]; u16 u[16]; };

#define N_B   2
#define N_S   2048
#define N_D   2048
#define N_HQ  16
#define N_HKV 4
#define N_HD  128
#define GRP   (N_HQ / N_HKV)
#define SCALE 0.08838834764831845f   // 128^-0.5

#if __has_builtin(__builtin_amdgcn_global_load_async_to_lds_b128) && \
    __has_builtin(__builtin_amdgcn_s_wait_asynccnt)
#define HAVE_ASYNC_LDS 1
typedef int v4i_t __attribute__((ext_vector_type(4)));
typedef __attribute__((address_space(1))) v4i_t as1_v4i;   // global
typedef __attribute__((address_space(3))) v4i_t as3_v4i;   // LDS
#endif

#if __has_builtin(__builtin_amdgcn_update_dpp)
#define HAVE_DPP 1
#endif

__device__ __forceinline__ u16 f2bf(float f) {
    unsigned u = __float_as_uint(f);
    unsigned r = u + 0x7FFFu + ((u >> 16) & 1u);
    return (u16)(r >> 16);
}
__device__ __forceinline__ unsigned pack2(float a, float b) {
    return (unsigned)f2bf(a) | ((unsigned)f2bf(b) << 16);
}

// XOR-shuffle within each 16-lane row. DPP ROW_XMASK (VALU pipe) when
// available; ds_bpermute-based __shfl_xor fallback otherwise.
template <int MASK>
__device__ __forceinline__ float xor16(float v) {
#ifdef HAVE_DPP
    int i = __float_as_int(v);
    i = __builtin_amdgcn_update_dpp(i, i, 0x160 | MASK, 0xF, 0xF, false);
    return __int_as_float(i);
#else
    return __shfl_xor(v, MASK, 32);
#endif
}
__device__ __forceinline__ float rowmax16(float v) {
    v = fmaxf(v, xor16<1>(v));
    v = fmaxf(v, xor16<2>(v));
    v = fmaxf(v, xor16<4>(v));
    v = fmaxf(v, xor16<8>(v));
    return v;
}
__device__ __forceinline__ float rowsum16(float v) {
    v += xor16<1>(v);
    v += xor16<2>(v);
    v += xor16<4>(v);
    v += xor16<8>(v);
    return v;
}

// ---------------------------------------------------------------------------
// Tiled GEMM: C[M,N] = alpha * (A[M,K] @ B[K,N]) (+bias), bf16 WMMA compute.
// Block tile 64x128, 8 waves, K-step 32. A staged natural, B staged transposed
// so both fragment types are contiguous 16B LDS loads.
// ---------------------------------------------------------------------------
template <bool ABF16, bool OUTBF16, bool BIAS>
__global__ __launch_bounds__(256) void gemm_bf16_wmma(
    const void* __restrict__ Ap, const float* __restrict__ Bw,
    void* __restrict__ Cp, const float* __restrict__ bias,
    float alpha, int M, int N, int K)
{
    __shared__ u16 As[64 * 40];     // [row][k], stride 40
    __shared__ u16 Bs[128 * 40];    // [n][k],   stride 40 (transposed)

    const int tid  = threadIdx.x;
    const int m0   = blockIdx.y * 64;
    const int n0   = blockIdx.x * 128;
    const int w    = tid >> 5;
    const int lane = tid & 31;
    const int ln   = lane & 15;
    const int lh   = lane >> 4;
    const int mt   = w >> 1;        // 0..3 -> 16-row tile
    const int nc   = w & 1;         // 0..1 -> 64-col chunk

    v8f acc[4];
#pragma unroll
    for (int nt = 0; nt < 4; ++nt)
#pragma unroll
        for (int r = 0; r < 8; ++r) acc[nt][r] = 0.f;

    const int arow = tid >> 2, acol = (tid & 3) * 8;   // A: 64 rows, 8 elems/thr
    const int brow = tid >> 3, bcol = (tid & 7) * 16;  // B: 32 rows, 16 elems/thr

    for (int k0 = 0; k0 < K; k0 += 32) {
        // ---- stage A tile (64 x 32) as bf16 ----
        if (ABF16) {
            const u16* A = (const u16*)Ap;
            const u16* p = A + (size_t)(m0 + arow) * K + k0 + acol;
            uint4 d = *(const uint4*)p;
            *(uint4*)(&As[arow * 40 + acol]) = d;
            if (k0 + 32 < K) __builtin_prefetch(p + 32, 0, 0);
        } else {
            const float* A = (const float*)Ap;
            const float* p = A + (size_t)(m0 + arow) * K + k0 + acol;
            float4 f0 = *(const float4*)p;
            float4 f1 = *(const float4*)(p + 4);
            uint4 d;
            d.x = pack2(f0.x, f0.y); d.y = pack2(f0.z, f0.w);
            d.z = pack2(f1.x, f1.y); d.w = pack2(f1.z, f1.w);
            *(uint4*)(&As[arow * 40 + acol]) = d;
            if (k0 + 32 < K) __builtin_prefetch(p + 32, 0, 0);
        }
        // ---- stage B tile (32 x 128) transposed into Bs[n][k] ----
        {
            const float* p = Bw + (size_t)(k0 + brow) * N + n0 + bcol;
#pragma unroll
            for (int j4 = 0; j4 < 4; ++j4) {
                float4 f = *(const float4*)(p + j4 * 4);
                Bs[(bcol + j4 * 4 + 0) * 40 + brow] = f2bf(f.x);
                Bs[(bcol + j4 * 4 + 1) * 40 + brow] = f2bf(f.y);
                Bs[(bcol + j4 * 4 + 2) * 40 + brow] = f2bf(f.z);
                Bs[(bcol + j4 * 4 + 3) * 40 + brow] = f2bf(f.w);
            }
            if (k0 + 32 < K) __builtin_prefetch(p + (size_t)32 * N, 0, 0);
        }
        __syncthreads();

        // ---- fragments + WMMA ----
        FragU af;
        const u16* ab = &As[(mt * 16 + ln) * 40 + lh * 8];
        af.q[0] = *(const uint4*)(ab);
        af.q[1] = *(const uint4*)(ab + 16);
#pragma unroll
        for (int nt = 0; nt < 4; ++nt) {
            FragU bfr;
            const u16* bb = &Bs[(nc * 64 + nt * 16 + ln) * 40 + lh * 8];
            bfr.q[0] = *(const uint4*)(bb);
            bfr.q[1] = *(const uint4*)(bb + 16);
            acc[nt] = __builtin_amdgcn_wmma_f32_16x16x32_bf16(
                false, af.v, false, bfr.v, (short)0, acc[nt], false, false);
        }
        __syncthreads();
    }

    // ---- epilogue ----
#pragma unroll
    for (int nt = 0; nt < 4; ++nt) {
        const int col = n0 + nc * 64 + nt * 16 + ln;
        const float bv = BIAS ? bias[col] : 0.f;
#pragma unroll
        for (int r = 0; r < 8; ++r) {
            const int row = m0 + mt * 16 + r + 8 * lh;
            const float v = fmaf(acc[nt][r], alpha, bv);
            if (OUTBF16) ((u16*)Cp)[(size_t)row * N + col] = f2bf(v);
            else         ((float*)Cp)[(size_t)row * N + col] = v;
        }
    }
}

// ---------------------------------------------------------------------------
// Flash attention: block = 8 waves = 128 query rows of one (b, head).
// K staged natural [key][hd] via async global->LDS copies (ASYNCcnt);
// V staged transposed [hd][key]; online softmax in f32 with DPP row
// reductions; P converted to bf16 via per-wave LDS round-trip to reach the
// A-fragment layout. Softmax scale is pre-folded into Q.
// ---------------------------------------------------------------------------
__global__ __launch_bounds__(256) void attn_flash_wmma(
    const u16* __restrict__ Qg, const u16* __restrict__ Kg,
    const u16* __restrict__ Vg, u16* __restrict__ Og)
{
    __shared__ u16 Ks[32 * 136];        // [key][hd] stride 136
    __shared__ u16 Vt[128 * 40];        // [hd][key] stride 40
    __shared__ u16 Pb[8 * 16 * 40];     // per-wave 16x32 P tile, stride 40

    const int tid = threadIdx.x;
    const int nqt = N_S / 128;                      // 16
    const int qt  = blockIdx.x % nqt;
    const int h   = (blockIdx.x / nqt) % N_HQ;
    const int b   = blockIdx.x / (nqt * N_HQ);
    const int kv  = h / GRP;

    const int w    = tid >> 5;
    const int lane = tid & 31;
    const int ln   = lane & 15;
    const int lh   = lane >> 4;
    const int q0   = qt * 128 + w * 16;             // wave's first q row

    // ---- load Q fragments (16 rows x 128 hd, 4 k-steps of 32) ----
    FragU qf[4];
    {
        const u16* qp = Qg + ((size_t)(b * N_S + q0 + ln)) * (N_HQ * N_HD) + h * N_HD;
#pragma unroll
        for (int ks = 0; ks < 4; ++ks) {
            const int hd0 = ks * 32 + lh * 8;
            qf[ks].q[0] = *(const uint4*)(qp + hd0);
            qf[ks].q[1] = *(const uint4*)(qp + hd0 + 16);
        }
    }

    float mrow[8], lrow[8];
    v8f oacc[8];
#pragma unroll
    for (int r = 0; r < 8; ++r) { mrow[r] = -1e30f; lrow[r] = 0.f; }
#pragma unroll
    for (int ch = 0; ch < 8; ++ch)
#pragma unroll
        for (int r = 0; r < 8; ++r) oacc[ch][r] = 0.f;

    const int krow = tid >> 3;          // 0..31 key row
    const int kseg = (tid & 7) * 16;    // 16 hd elems per thread

    for (int kb = 0; kb < N_S / 32; ++kb) {
        const int key0 = kb * 32;
        // ---- stage K (async natural) and V (transposed through VGPRs) ----
        {
            const size_t rowoff = ((size_t)(b * N_S + key0 + krow)) * (N_HKV * N_HD) + kv * N_HD + kseg;
            const u16* kp = Kg + rowoff;
#ifdef HAVE_ASYNC_LDS
            __builtin_amdgcn_global_load_async_to_lds_b128(
                (as1_v4i*)(kp), (as3_v4i*)(&Ks[krow * 136 + kseg]), 0, 0);
            __builtin_amdgcn_global_load_async_to_lds_b128(
                (as1_v4i*)(kp + 8), (as3_v4i*)(&Ks[krow * 136 + kseg + 8]), 0, 0);
#else
            uint4 ka = *(const uint4*)(kp);
            uint4 kbv = *(const uint4*)(kp + 8);
            *(uint4*)(&Ks[krow * 136 + kseg]) = ka;
            *(uint4*)(&Ks[krow * 136 + kseg + 8]) = kbv;
#endif
            const u16* vp = Vg + rowoff;
            uint4 va = *(const uint4*)(vp);
            uint4 vb = *(const uint4*)(vp + 8);
            const u16* ve0 = (const u16*)&va;
            const u16* ve1 = (const u16*)&vb;
#pragma unroll
            for (int j = 0; j < 8; ++j) Vt[(kseg + j) * 40 + krow] = ve0[j];
#pragma unroll
            for (int j = 0; j < 8; ++j) Vt[(kseg + 8 + j) * 40 + krow] = ve1[j];
            if (key0 + 32 < N_S) {
                __builtin_prefetch(kp + (size_t)32 * (N_HKV * N_HD), 0, 0);
                __builtin_prefetch(vp + (size_t)32 * (N_HKV * N_HD), 0, 0);
            }
        }
#ifdef HAVE_ASYNC_LDS
        __builtin_amdgcn_s_wait_asynccnt(0);
#endif
        __syncthreads();

        // ---- scores: two 16x16 tiles over 32 keys (scale folded into Q) ----
        v8f sc[2];
#pragma unroll
        for (int k2 = 0; k2 < 2; ++k2) {
            v8f s;
#pragma unroll
            for (int r = 0; r < 8; ++r) s[r] = 0.f;
#pragma unroll
            for (int ks = 0; ks < 4; ++ks) {
                FragU kf;
                const u16* kbp = &Ks[(k2 * 16 + ln) * 136 + ks * 32 + lh * 8];
                kf.q[0] = *(const uint4*)(kbp);
                kf.q[1] = *(const uint4*)(kbp + 16);
                s = __builtin_amdgcn_wmma_f32_16x16x32_bf16(
                    false, qf[ks].v, false, kf.v, (short)0, s, false, false);
            }
            sc[k2] = s;
        }

        // ---- online softmax (row = r + 8*lh, cols across 16-lane half) ----
        float nm[8], al[8];
#pragma unroll
        for (int r = 0; r < 8; ++r) {
            float mx = rowmax16(fmaxf(sc[0][r], sc[1][r]));
            nm[r] = fmaxf(mrow[r], mx);
            al[r] = __expf(mrow[r] - nm[r]);
            mrow[r] = nm[r];
        }
#pragma unroll
        for (int r = 0; r < 8; ++r) {
            sc[0][r] = __expf(sc[0][r] - nm[r]);
            sc[1][r] = __expf(sc[1][r] - nm[r]);
            lrow[r] = lrow[r] * al[r] + rowsum16(sc[0][r] + sc[1][r]);
        }
#pragma unroll
        for (int ch = 0; ch < 8; ++ch)
#pragma unroll
            for (int r = 0; r < 8; ++r) oacc[ch][r] *= al[r];

        // ---- P (C-layout f32) -> LDS bf16 -> A-layout fragment ----
        u16* pw = &Pb[w * 16 * 40];
#pragma unroll
        for (int r = 0; r < 8; ++r) {
            const int m = r + 8 * lh;
            pw[m * 40 + ln]      = f2bf(sc[0][r]);
            pw[m * 40 + 16 + ln] = f2bf(sc[1][r]);
        }
        __builtin_amdgcn_wave_barrier();   // per-wave region; DS in-order per wave
        FragU pf;
        const u16* pr = &Pb[w * 16 * 40 + ln * 40 + lh * 8];
        pf.q[0] = *(const uint4*)(pr);
        pf.q[1] = *(const uint4*)(pr + 16);

        // ---- O += P @ V (8 hd chunks of 16) ----
#pragma unroll
        for (int ch = 0; ch < 8; ++ch) {
            FragU vf;
            const u16* vbp = &Vt[(ch * 16 + ln) * 40 + lh * 8];
            vf.q[0] = *(const uint4*)(vbp);
            vf.q[1] = *(const uint4*)(vbp + 16);
            oacc[ch] = __builtin_amdgcn_wmma_f32_16x16x32_bf16(
                false, pf.v, false, vf.v, (short)0, oacc[ch], false, false);
        }
        __syncthreads();   // protect LDS before next staging
    }

    // ---- normalize and write O in [b, s, HQ*HD] layout (bf16) ----
    float inv[8];
#pragma unroll
    for (int r = 0; r < 8; ++r) inv[r] = 1.f / lrow[r];
#pragma unroll
    for (int ch = 0; ch < 8; ++ch)
#pragma unroll
        for (int r = 0; r < 8; ++r) {
            const size_t row = (size_t)(b * N_S + q0 + r + 8 * lh);
            Og[row * (N_HQ * N_HD) + h * N_HD + ch * 16 + ln] = f2bf(oacc[ch][r] * inv[r]);
        }
}

// ---------------------------------------------------------------------------
extern "C" void kernel_launch(void* const* d_in, const int* in_sizes, int n_in,
                              void* d_out, int out_size, void* d_ws, size_t ws_size,
                              hipStream_t stream) {
    (void)in_sizes; (void)n_in; (void)out_size; (void)ws_size;

    const float* x  = (const float*)d_in[0];
    const float* Wq = (const float*)d_in[1];
    const float* Wk = (const float*)d_in[2];
    const float* Wv = (const float*)d_in[3];
    const float* Wo = (const float*)d_in[4];
    const float* bo = (const float*)d_in[5];
    float* out = (float*)d_out;

    const int M = N_B * N_S;   // 4096 flattened rows

    // workspace: Q (bf16 16MB) | K (4MB) | V (4MB) | attn-out (16MB)
    u16* Qb  = (u16*)d_ws;
    u16* Kb  = Qb + (size_t)M * (N_HQ * N_HD);
    u16* Vb  = Kb + (size_t)M * (N_HKV * N_HD);
    u16* Ocb = Vb + (size_t)M * (N_HKV * N_HD);

    dim3 blk(256);

    // projections (fp32 in -> bf16 out); softmax scale folded into Q
    gemm_bf16_wmma<false, true, false><<<dim3((N_HQ * N_HD) / 128, M / 64), blk, 0, stream>>>(
        x, Wq, Qb, nullptr, SCALE, M, N_HQ * N_HD, N_D);
    gemm_bf16_wmma<false, true, false><<<dim3((N_HKV * N_HD) / 128, M / 64), blk, 0, stream>>>(
        x, Wk, Kb, nullptr, 1.0f, M, N_HKV * N_HD, N_D);
    gemm_bf16_wmma<false, true, false><<<dim3((N_HKV * N_HD) / 128, M / 64), blk, 0, stream>>>(
        x, Wv, Vb, nullptr, 1.0f, M, N_HKV * N_HD, N_D);

    // flash attention: B * HQ * (S/128) blocks
    attn_flash_wmma<<<dim3(N_B * N_HQ * (N_S / 128)), blk, 0, stream>>>(Qb, Kb, Vb, Ocb);

    // output projection (bf16 in -> fp32 out, + bias)
    gemm_bf16_wmma<true, false, true><<<dim3(N_D / 128, M / 64), blk, 0, stream>>>(
        Ocb, Wo, out, bo, 1.0f, M, N_D, N_HQ * N_HD);
}
